// VisionMamba_4776003633364
// MI455X (gfx1250) — matrix operand order repaired and checked by reading, not compiled, not run on hardware
//
#include <hip/hip_runtime.h>
#include <hip/hip_bf16.h>
#include <math.h>

#define E_     384
#define DIN_   768
#define NS_    16
#define KC_    4
#define RR_    24
#define DEPTH_ 12
#define BB_    2
#define LL_    196
#define MM_    (BB_*LL_)   /* 392 rows (b,l) */

typedef __attribute__((ext_vector_type(16))) __bf16         v16bf;
typedef __attribute__((ext_vector_type(2)))  __bf16         v2bf;
typedef __attribute__((ext_vector_type(16))) unsigned short v16u;
typedef __attribute__((ext_vector_type(8)))  float          v8f;

union BfCvt { v16u u; v16bf b; };

#if defined(__has_builtin)
#  if __has_builtin(__builtin_amdgcn_cvt_pk_bf16_f32)
#    define HAVE_CVT_PK_BF16 1
#  endif
#endif

__device__ __forceinline__ unsigned short f2bf(float f) {
  unsigned int u = __float_as_uint(f);
  u += 0x7FFFu + ((u >> 16) & 1u);            // RNE truncate to bf16
  return (unsigned short)(u >> 16);
}
__device__ __forceinline__ unsigned int pack2bf(float a, float b) {
#ifdef HAVE_CVT_PK_BF16
  v2bf r = __builtin_amdgcn_cvt_pk_bf16_f32(a, b);   // v_cvt_pk_bf16_f32 (RNE)
  return __builtin_bit_cast(unsigned int, r);
#else
  return (unsigned int)f2bf(a) | ((unsigned int)f2bf(b) << 16);
#endif
}
__device__ __forceinline__ float softplus_(float x) {
  return (x > 20.f) ? x : log1pf(__expf(x));
}
__device__ __forceinline__ float silu_(float x) {
  return x / (1.f + __expf(-x));
}

// ---------------------------------------------------------------------------
// bf16 WMMA GEMM:  C[M,Nc] = A[M,Kc] * W[Nc,Kc]^T (+bias)
// Optional flip-fuse: A_eff[m,k] = A[m,k] + A2[(b*L + (L-1-l))*lda + k]
// Block = 128 threads (4 waves); block tile 32(M) x 64(N); K step 32.
// Each wave: two accumulators sharing one B fragment -> 2 WMMA per K-step.
// Requires: lda % 4 == 0, Kc % 4 == 0, 16B-aligned A/A2/W (true for all calls).
// ---------------------------------------------------------------------------
__global__ void gemm_wmma(const float* __restrict__ A, const float* __restrict__ A2,
                          const float* __restrict__ W, const float* __restrict__ bias,
                          float* __restrict__ C,
                          int M, int Nc, int Kc, int lda, int Lflip) {
  __shared__ unsigned short sA[32][32];
  __shared__ unsigned short sW[64][32];

  const int tidx = threadIdx.x;
  const int wave = tidx >> 5;
  const int lane = tidx & 31;
  const int m0 = blockIdx.x * 32;
  const int n0 = blockIdx.y * 64;

  v8f acc0 = {0.f,0.f,0.f,0.f,0.f,0.f,0.f,0.f};
  v8f acc1 = {0.f,0.f,0.f,0.f,0.f,0.f,0.f,0.f};

  for (int k0 = 0; k0 < Kc; k0 += 32) {
    // stage A tile (32x32) as bf16 via 128-bit loads: 256 float4-groups, 2/thread
#pragma unroll
    for (int gg = 0; gg < 2; ++gg) {
      int g = tidx + gg * 128;
      int mi = g >> 3, kg = (g & 7) * 4;
      int m = m0 + mi, k = k0 + kg;
      float4 v = make_float4(0.f, 0.f, 0.f, 0.f);
      if (m < M && k < Kc) {
        v = *(const float4*)(A + (size_t)m * lda + k);
        if (Lflip) {
          int b = m / Lflip, l = m % Lflip;
          float4 v2 = *(const float4*)(A2 + (size_t)(b * Lflip + (Lflip - 1 - l)) * lda + k);
          v.x += v2.x; v.y += v2.y; v.z += v2.z; v.w += v2.w;
        }
      }
      uint2 p; p.x = pack2bf(v.x, v.y); p.y = pack2bf(v.z, v.w);
      *(uint2*)&sA[mi][kg] = p;
    }
    // stage W tile (64x32) as bf16: 512 float4-groups, 4/thread
#pragma unroll
    for (int gg = 0; gg < 4; ++gg) {
      int g = tidx + gg * 128;
      int ni = g >> 3, kg = (g & 7) * 4;
      int n = n0 + ni, k = k0 + kg;
      float4 v = make_float4(0.f, 0.f, 0.f, 0.f);
      if (n < Nc && k < Kc) v = *(const float4*)(W + (size_t)n * Kc + k);
      uint2 p; p.x = pack2bf(v.x, v.y); p.y = pack2bf(v.z, v.w);
      *(uint2*)&sW[ni][kg] = p;
    }
    __syncthreads();

    // pack fragments per ISA 7.12.2 wave32 layouts
    BfCvt a0, a1, bfr;
    const int mrow = lane & 15;
    const int hi   = (lane >> 4) & 1;       // lane[4]
#pragma unroll
    for (int h = 0; h < 16; ++h) {
      int kk = h + ((h >= 8) ? 8 : 0) + hi * 8;   // A 16x32 K mapping
      a0.u[h] = sA[mrow][kk];
      a1.u[h] = sA[mrow + 16][kk];
    }
    const int ncol = (lane & 15) + wave * 16;
#pragma unroll
    for (int h = 0; h < 16; ++h) {
      bfr.u[h] = sW[ncol][h + ((h >= 8) ? 8 : 0) + hi * 8];           // B 32x16 K mapping
    }
    acc0 = __builtin_amdgcn_wmma_f32_16x16x32_bf16(
        false, a0.b, false, bfr.b, (short)0, acc0, false, false);
    acc1 = __builtin_amdgcn_wmma_f32_16x16x32_bf16(
        false, a1.b, false, bfr.b, (short)0, acc1, false, false);
    __syncthreads();
  }

  // C/D layout: vgpr r -> M = r + 8*lane[4]; N = lane&15
  const int nout = n0 + wave * 16 + (lane & 15);
  if (nout < Nc) {
    const float bv = bias ? bias[nout] : 0.f;
    const int mb0 = m0 + ((lane >> 4) ? 8 : 0);
#pragma unroll
    for (int r = 0; r < 8; ++r) {
      int m = mb0 + r;
      if (m < M) C[(size_t)m * Nc + nout] = acc0[r] + bv;
    }
#pragma unroll
    for (int r = 0; r < 8; ++r) {
      int m = mb0 + 16 + r;
      if (m < M) C[(size_t)m * Nc + nout] = acc1[r] + bv;
    }
  }
}

// ---------------------------------------------------------------------------
// Patch embed: res[m,e] = patch(m) . patch_w[e] + patch_b[e] + pos[l,e]
// ---------------------------------------------------------------------------
__global__ void patch_embed(const float* __restrict__ img, const float* __restrict__ pw,
                            const float* __restrict__ pb, const float* __restrict__ pos,
                            float* __restrict__ res) {
  int m = blockIdx.x;
  int e = blockIdx.y * 128 + threadIdx.x;
  if (e >= E_) return;
  int b = m / LL_, l = m % LL_;
  int py = l / 14, px = l % 14;
  const float* ip = img + (size_t)b * 224 * 224 + (py * 16) * 224 + px * 16;
  const float* wp = pw + (size_t)e * 256;
  float s = 0.f;
#pragma unroll 4
  for (int iy = 0; iy < 16; ++iy)
    for (int ix = 0; ix < 16; ++ix)
      s += ip[iy * 224 + ix] * wp[iy * 16 + ix];
  res[(size_t)m * E_ + e] = s + pb[e] + pos[(size_t)l * E_ + e];
}

// ---------------------------------------------------------------------------
// res (+= hid) ; out = rms(res)*w      (one row per block, 128 threads)
// ---------------------------------------------------------------------------
__global__ void res_rms(float* __restrict__ res, const float* __restrict__ hid,
                        const float* __restrict__ w, float* __restrict__ out, int addhid) {
  __shared__ float red[128];
  int row = blockIdx.x, tid = threadIdx.x;
  float v[3]; float ss = 0.f;
#pragma unroll
  for (int i = 0; i < 3; ++i) {
    int j = tid + i * 128;
    float x = res[(size_t)row * E_ + j];
    if (addhid) { x += hid[(size_t)row * E_ + j]; res[(size_t)row * E_ + j] = x; }
    v[i] = x; ss += x * x;
  }
  red[tid] = ss; __syncthreads();
  for (int s = 64; s > 0; s >>= 1) { if (tid < s) red[tid] += red[tid + s]; __syncthreads(); }
  float scale = rsqrtf(red[0] / (float)E_ + 1e-5f);
#pragma unroll
  for (int i = 0; i < 3; ++i) {
    int j = tid + i * 128;
    out[(size_t)row * E_ + j] = v[i] * scale * w[j];
  }
}

// ---------------------------------------------------------------------------
// Causal depthwise conv (K=4) + SiLU, both directions.
// xz: [b,l,1536] (xh = cols 0..767).  xc: [dir,b,l,768] in scan order.
// ---------------------------------------------------------------------------
__global__ void dwconv_silu(const float* __restrict__ xz,
                            const float* __restrict__ cw, const float* __restrict__ cb,
                            const float* __restrict__ cwr, const float* __restrict__ cbr,
                            float* __restrict__ xc) {
  int idx = blockIdx.x * 256 + threadIdx.x;
  const int total = 2 * BB_ * LL_ * DIN_;
  if (idx >= total) return;
  int d = idx % DIN_; int t = idx / DIN_;
  int l = t % LL_; t /= LL_;
  int b = t % BB_; int dir = t / BB_;
  const float* w  = dir ? cwr : cw;
  float acc = (dir ? cbr : cb)[d];
#pragma unroll
  for (int j = 0; j < KC_; ++j) {
    int p = l - (KC_ - 1) + j;
    if (p >= 0) {
      int s = dir ? (LL_ - 1 - p) : p;
      acc += w[d * KC_ + j] * xz[(size_t)(b * LL_ + s) * (2 * DIN_) + d];
    }
  }
  xc[(size_t)((dir * BB_ + b) * LL_ + l) * DIN_ + d] = silu_(acc);
}

// ---------------------------------------------------------------------------
// Selective scan: one thread per (dir,b,d); 16-wide state in registers.
// dt = softplus(dtraw + dtb); h = exp(dt*A)h + (dt*u)B; y = <h,C>;
// out = (y + u*D)*silu(z)  ->  ybuf[dir,b,l,d] (scan-local order).
// ---------------------------------------------------------------------------
__global__ void ssm_scan(const float* __restrict__ dtraw, const float* __restrict__ xc,
                         const float* __restrict__ dbl, const float* __restrict__ xz,
                         float* __restrict__ ybuf,
                         const float* __restrict__ Alog0, const float* __restrict__ Alogr,
                         const float* __restrict__ D0, const float* __restrict__ Dr,
                         const float* __restrict__ dtb0, const float* __restrict__ dtbr) {
  int tid = blockIdx.x * blockDim.x + threadIdx.x;
  if (tid >= 2 * BB_ * DIN_) return;
  int d = tid % DIN_;
  int b = (tid / DIN_) % BB_;
  int dir = tid / (DIN_ * BB_);
  const float* Al = (dir ? Alogr : Alog0) + (size_t)d * NS_;
  float An[NS_];
#pragma unroll
  for (int n = 0; n < NS_; ++n) An[n] = -__expf(Al[n]);
  const float Dv   = (dir ? Dr : D0)[d];
  const float dtbv = (dir ? dtbr : dtb0)[d];
  float h[NS_];
#pragma unroll
  for (int n = 0; n < NS_; ++n) h[n] = 0.f;

  for (int l = 0; l < LL_; ++l) {
    size_t rix = (size_t)((dir * BB_ + b) * LL_ + l);
    float dtv = softplus_(dtraw[rix * DIN_ + d] + dtbv);
    float u   = xc[rix * DIN_ + d];
    const float* dr = dbl + rix * (RR_ + 2 * NS_);
    float dtu = dtv * u;
    float y = 0.f;
#pragma unroll
    for (int n = 0; n < NS_; ++n) {
      h[n] = __expf(dtv * An[n]) * h[n] + dtu * dr[RR_ + n];
      y += h[n] * dr[RR_ + NS_ + n];
    }
    int zl = dir ? (LL_ - 1 - l) : l;
    float z = xz[(size_t)(b * LL_ + zl) * (2 * DIN_) + DIN_ + d];
    ybuf[rix * DIN_ + d] = (y + u * Dv) * silu_(z);
  }
}

// ---------------------------------------------------------------------------
extern "C" void kernel_launch(void* const* d_in, const int* in_sizes, int n_in,
                              void* d_out, int out_size, void* d_ws, size_t ws_size,
                              hipStream_t stream) {
  const float* x       = (const float*)d_in[0];
  const float* patch_w = (const float*)d_in[1];
  const float* patch_b = (const float*)d_in[2];
  const float* pos     = (const float*)d_in[3];
  const float* norm_w  = (const float*)d_in[4];
  const float* ipw     = (const float*)d_in[5];
  const float* cw      = (const float*)d_in[6];
  const float* cb      = (const float*)d_in[7];
  const float* xpw     = (const float*)d_in[8];
  const float* dtw     = (const float*)d_in[9];
  const float* dtb     = (const float*)d_in[10];
  const float* Alog    = (const float*)d_in[11];
  const float* Dp      = (const float*)d_in[12];
  const float* cwr     = (const float*)d_in[13];
  const float* cbr     = (const float*)d_in[14];
  const float* xpwr    = (const float*)d_in[15];
  const float* dtwr    = (const float*)d_in[16];
  const float* dtbr    = (const float*)d_in[17];
  const float* Alogr   = (const float*)d_in[18];
  const float* Dpr     = (const float*)d_in[19];
  const float* opw     = (const float*)d_in[20];
  const float* normf   = (const float*)d_in[21];

  const int NX = RR_ + 2 * NS_;  // 56
  float* ws = (float*)d_ws;
  float* res   = ws; ws += (size_t)MM_ * E_;
  float* hid   = ws; ws += (size_t)MM_ * E_;
  float* xn    = ws; ws += (size_t)MM_ * E_;
  float* xz    = ws; ws += (size_t)MM_ * 2 * DIN_;
  float* xc    = ws; ws += (size_t)2 * MM_ * DIN_;
  float* dbl   = ws; ws += (size_t)2 * MM_ * NX;
  float* dtraw = ws; ws += (size_t)2 * MM_ * DIN_;
  float* ybuf  = ws; ws += (size_t)2 * MM_ * DIN_;

  patch_embed<<<dim3(MM_, 3), 128, 0, stream>>>(x, patch_w, patch_b, pos, res);

  for (int i = 0; i < DEPTH_; ++i) {
    // res (+= hid), xn = rms(res)*norm_w[i]
    res_rms<<<MM_, 128, 0, stream>>>(res, hid, norm_w + (size_t)i * E_, xn, i > 0 ? 1 : 0);

    // in_proj: xz[392,1536] = xn[392,384] * ipw^T
    gemm_wmma<<<dim3((MM_ + 31) / 32, (2 * DIN_ + 63) / 64), 128, 0, stream>>>(
        xn, nullptr, ipw + (size_t)i * 2 * DIN_ * E_, nullptr, xz,
        MM_, 2 * DIN_, E_, E_, 0);

    // depthwise conv + silu, both directions
    dwconv_silu<<<(2 * BB_ * LL_ * DIN_ + 255) / 256, 256, 0, stream>>>(
        xz, cw + (size_t)i * DIN_ * KC_, cb + (size_t)i * DIN_,
        cwr + (size_t)i * DIN_ * KC_, cbr + (size_t)i * DIN_, xc);

    for (int dir = 0; dir < 2; ++dir) {
      const float* xp = (dir ? xpwr : xpw) + (size_t)i * NX * DIN_;
      const float* dw = (dir ? dtwr : dtw) + (size_t)i * DIN_ * RR_;
      // x_proj: dbl[392,56] = xc_dir[392,768] * xp^T
      gemm_wmma<<<dim3((MM_ + 31) / 32, 1), 128, 0, stream>>>(
          xc + (size_t)dir * MM_ * DIN_, nullptr, xp, nullptr,
          dbl + (size_t)dir * MM_ * NX, MM_, NX, DIN_, DIN_, 0);
      // dt_proj: dtraw[392,768] = dbl[:, :24] * dw^T   (K=24 zero-padded to 32)
      gemm_wmma<<<dim3((MM_ + 31) / 32, (DIN_ + 63) / 64), 128, 0, stream>>>(
          dbl + (size_t)dir * MM_ * NX, nullptr, dw, nullptr,
          dtraw + (size_t)dir * MM_ * DIN_, MM_, DIN_, RR_, NX, 0);
    }

    // bidirectional selective scan (softplus + dt bias fused)
    ssm_scan<<<(2 * BB_ * DIN_ + 127) / 128, 128, 0, stream>>>(
        dtraw, xc, dbl, xz, ybuf,
        Alog + (size_t)i * DIN_ * NS_, Alogr + (size_t)i * DIN_ * NS_,
        Dp + (size_t)i * DIN_, Dpr + (size_t)i * DIN_,
        dtb + (size_t)i * DIN_, dtbr + (size_t)i * DIN_);

    // out_proj with fused flip-add of the backward pass:
    // hid[392,384] = (yf + flip(yb)) * opw^T
    gemm_wmma<<<dim3((MM_ + 31) / 32, (E_ + 63) / 64), 128, 0, stream>>>(
        ybuf, ybuf + (size_t)MM_ * DIN_, opw + (size_t)i * E_ * DIN_, nullptr,
        hid, MM_, E_, DIN_, DIN_, LL_);
  }

  // final: out = rms(res + hid) * normf_w
  res_rms<<<MM_, 128, 0, stream>>>(res, hid, normf, (float*)d_out, 1);
}
